// Net_screen_DTI_85478439125041
// MI455X (gfx1250) — compile-verified, matrix operand order checked by reading, not run on hardware
//
#include <hip/hip_runtime.h>
#include <math.h>

#define NN 100000
#define EE 600000
#define DD 128
#define EDD 3
#define DFF 512
#define BB 64
#define CC 2
#define NLAYERS 4   // conv1 + 3 stacked
#define MTILES 5    // 80 rows per GEMM block (100000 = 1250 * 80)

typedef __attribute__((ext_vector_type(16))) __bf16 v16bf;
typedef __attribute__((ext_vector_type(8)))  __bf16 v8bf;
typedef __attribute__((ext_vector_type(8)))  float  v8f;

__device__ __forceinline__ unsigned short f2bf(float f) {
  unsigned int u = __float_as_uint(f);
  u += 0x7FFFu + ((u >> 16) & 1u);   // round-to-nearest-even
  return (unsigned short)(u >> 16);
}

// ---------------------------------------------------------------------------
// f32 -> bf16 conversion
// ---------------------------------------------------------------------------
__global__ void k_f32_to_bf16(const float* __restrict__ in,
                              unsigned short* __restrict__ out, int n) {
  int i = blockIdx.x * blockDim.x + threadIdx.x;
  if (i < n) out[i] = f2bf(in[i]);
}

// ---------------------------------------------------------------------------
// WMMA GEMM: out[M,128] = A[M,128](bf16) @ W[128,128]^T(bf16, row-major [n][k]) + bias
// block = 256 threads = 8 waves; block computes 80 rows x 128 cols.
// Wave w owns cols [16w,16w+16): loads its 4 B fragments ONCE into registers,
// then sweeps 5 M-tiles from the LDS-staged A tile -> 20 WMMAs per wave.
// ---------------------------------------------------------------------------
__global__ __launch_bounds__(256)
void k_gemm_bf16_128(const unsigned short* __restrict__ A,
                     const unsigned short* __restrict__ W,
                     const float* __restrict__ bias,
                     float* __restrict__ out, int M) {
  __shared__ unsigned short As[MTILES * 16 * 128];       // 20KB
  const int tid   = threadIdx.x;
  const int mbase = blockIdx.x * (MTILES * 16);
  // stage A tile: 20480 bytes = 1280 uint4 = 256 threads * 5
  {
    const uint4* srcp = (const uint4*)(A + (size_t)mbase * DD);
    uint4* dstp = (uint4*)As;
#pragma unroll
    for (int i = 0; i < MTILES; ++i) dstp[tid + 256 * i] = srcp[tid + 256 * i];
  }
  __syncthreads();

  const int wave  = tid >> 5;
  const int lane  = tid & 31;
  const int ncol  = wave * 16;
  const int mr    = lane & 15;          // A row in tile / B column
  const int khalf = (lane >> 4) * 8;    // lanes 0-15: K 0-7/16-23; 16-31: 8-15/24-31

  // B fragments for this wave's 16 columns, all of K=128, held in registers
  const unsigned short* wrow = W + (size_t)(ncol + mr) * DD;   // W[n][k], k contig
  v16bf bf[4];
#pragma unroll
  for (int kk = 0; kk < 4; ++kk) {
    const int kb = kk * 32;
    v8bf b0 = *(const v8bf*)(wrow + kb + khalf);
    v8bf b1 = *(const v8bf*)(wrow + kb + 16 + khalf);
    bf[kk] = __builtin_shufflevector(b0, b1, 0,1,2,3,4,5,6,7,8,9,10,11,12,13,14,15);
  }

  const int col  = ncol + mr;
  const float bv = bias[col];
#pragma unroll
  for (int mt = 0; mt < MTILES; ++mt) {
    const unsigned short* arow = As + (mt * 16 + mr) * DD;
    v8f c = {};
#pragma unroll
    for (int kk = 0; kk < 4; ++kk) {
      const int kb = kk * 32;
      v8bf a0 = *(const v8bf*)(arow + kb + khalf);
      v8bf a1 = *(const v8bf*)(arow + kb + 16 + khalf);
      v16bf a = __builtin_shufflevector(a0, a1, 0,1,2,3,4,5,6,7,8,9,10,11,12,13,14,15);
      c = __builtin_amdgcn_wmma_f32_16x16x32_bf16(false, a, false, bf[kk], (short)0, c,
                                                  false, false);
    }
    const int rbase = mbase + mt * 16 + ((lane >> 4) ? 8 : 0);  // C/D layout
#pragma unroll
    for (int i = 0; i < 8; ++i)
      out[(size_t)(rbase + i) * DD + col] = c[i] + bv;
  }
}

// ---------------------------------------------------------------------------
// init m[] = -inf
// ---------------------------------------------------------------------------
__global__ void k_init_neginf(float* __restrict__ m, int n) {
  int i = blockIdx.x * blockDim.x + threadIdx.x;
  if (i < n) m[i] = -INFINITY;
}

// ---------------------------------------------------------------------------
// logits[e] = scale * dot(q[dst], k[src] + edge_attr[e] @ we^T)
// one wave per edge, 4 features per lane, we (128x3) cached in LDS
// ---------------------------------------------------------------------------
__global__ __launch_bounds__(256)
void k_edge_logits(const int* __restrict__ src, const int* __restrict__ dst,
                   const float* __restrict__ attr, const float* __restrict__ we,
                   const float* __restrict__ q, const float* __restrict__ k,
                   float* __restrict__ logits, int E) {
  __shared__ float wes[DD * EDD];
  for (int i = threadIdx.x; i < DD * EDD; i += 256) wes[i] = we[i];
  __syncthreads();
  const int wave = threadIdx.x >> 5, lane = threadIdx.x & 31;
  const int e = blockIdx.x * 8 + wave;
  if (e >= E) return;
  const int s = src[e], d = dst[e];
  const float a0 = attr[e * 3], a1 = attr[e * 3 + 1], a2 = attr[e * 3 + 2];
  const float4 q4 = *(const float4*)(q + (size_t)d * DD + lane * 4);
  const float4 k4 = *(const float4*)(k + (size_t)s * DD + lane * 4);
  float p = 0.f;
#pragma unroll
  for (int j = 0; j < 4; ++j) {
    const int f = lane * 4 + j;
    const float ev = a0 * wes[f * 3] + a1 * wes[f * 3 + 1] + a2 * wes[f * 3 + 2];
    p += ((const float*)&q4)[j] * (((const float*)&k4)[j] + ev);
  }
#pragma unroll
  for (int msk = 16; msk >= 1; msk >>= 1) p += __shfl_xor(p, msk, 32);
  if (lane == 0) logits[e] = p * 0.08838834764831845f;   // 1/sqrt(128)
}

// segment max via float-as-int atomic ordering trick
__global__ void k_seg_max(const int* __restrict__ dst, const float* __restrict__ logits,
                          float* __restrict__ m, int E) {
  int i = blockIdx.x * blockDim.x + threadIdx.x;
  if (i >= E) return;
  const float v = logits[i];
  float* addr = m + dst[i];
  if (v >= 0.f) atomicMax((int*)addr, __float_as_int(v));
  else          atomicMin((unsigned int*)addr, __float_as_uint(v));
}

__global__ void k_seg_expsum(const int* __restrict__ dst, const float* __restrict__ logits,
                             const float* __restrict__ m, float* __restrict__ wexp,
                             float* __restrict__ den, int E) {
  int i = blockIdx.x * blockDim.x + threadIdx.x;
  if (i >= E) return;
  const int d = dst[i];
  const float w = __expf(logits[i] - m[d]);
  wexp[i] = w;
  atomicAdd(den + d, w);
}

// out[dst] += alpha * (v[src] + e);  one wave per edge
__global__ __launch_bounds__(256)
void k_edge_scatter(const int* __restrict__ src, const int* __restrict__ dst,
                    const float* __restrict__ attr, const float* __restrict__ we,
                    const float* __restrict__ v, const float* __restrict__ wexp,
                    const float* __restrict__ den, float* __restrict__ out, int E) {
  __shared__ float wes[DD * EDD];
  for (int i = threadIdx.x; i < DD * EDD; i += 256) wes[i] = we[i];
  __syncthreads();
  const int wave = threadIdx.x >> 5, lane = threadIdx.x & 31;
  const int e = blockIdx.x * 8 + wave;
  if (e >= E) return;
  const int s = src[e], d = dst[e];
  const float alpha = wexp[e] / (den[d] + 1e-16f);
  const float a0 = attr[e * 3], a1 = attr[e * 3 + 1], a2 = attr[e * 3 + 2];
  const float4 v4 = *(const float4*)(v + (size_t)s * DD + lane * 4);
#pragma unroll
  for (int j = 0; j < 4; ++j) {
    const int f = lane * 4 + j;
    const float ev = a0 * wes[f * 3] + a1 * wes[f * 3 + 1] + a2 * wes[f * 3 + 2];
    atomicAdd(out + (size_t)d * DD + f, alpha * (((const float*)&v4)[j] + ev));
  }
}

// h = relu(out2+skip) - relu(out1+skip); also emit bf16 copy for next layer GEMMs
__global__ void k_combine(const float* __restrict__ out1, const float* __restrict__ out2,
                          const float* __restrict__ skip, float* __restrict__ h,
                          unsigned short* __restrict__ hbf, size_t n) {
  size_t i = (size_t)blockIdx.x * blockDim.x + threadIdx.x;
  if (i >= n) return;
  const float s = skip[i];
  const float hv = fmaxf(out2[i] + s, 0.f) - fmaxf(out1[i] + s, 0.f);
  h[i] = hv;
  hbf[i] = f2bf(hv);
}

// ---------------------------------------------------------------------------
// pooling + MLP tail
// ---------------------------------------------------------------------------
__global__ void k_pool_sum(const float* __restrict__ h, const int* __restrict__ batchs,
                           float* __restrict__ sums, float* __restrict__ cnt, int N) {
  int i = blockIdx.x * blockDim.x + threadIdx.x;
  if (i >= N * DD) return;
  const int n = i >> 7, d = i & 127;
  const int b = batchs[n];
  atomicAdd(sums + b * DD + d, h[i]);
  if (d == 0) atomicAdd(cnt + b, 1.0f);
}

__global__ void k_pool_div(const float* __restrict__ sums, const float* __restrict__ cnt,
                           float* __restrict__ g, int n) {
  int i = blockIdx.x * blockDim.x + threadIdx.x;
  if (i < n) g[i] = sums[i] / fmaxf(cnt[i >> 7], 1.0f);
}

__global__ void k_fc(const float* __restrict__ in, const float* __restrict__ w,
                     const float* __restrict__ b, float* __restrict__ out,
                     int Bn, int ID, int OD, int relu) {
  int i = blockIdx.x * blockDim.x + threadIdx.x;
  if (i >= Bn * OD) return;
  const int bb = i / OD, o = i % OD;
  float acc = b[o];
  const float* ip = in + (size_t)bb * ID;
  const float* wp = w + (size_t)o * ID;
  for (int k2 = 0; k2 < ID; ++k2) acc += ip[k2] * wp[k2];
  out[i] = relu ? fmaxf(acc, 0.f) : acc;
}

__global__ void k_logsoftmax2(const float* __restrict__ lg, float* __restrict__ out, int Bn) {
  int b = blockIdx.x * blockDim.x + threadIdx.x;
  if (b >= Bn) return;
  const float a = lg[b * 2], c = lg[b * 2 + 1];
  const float mx = fmaxf(a, c);
  const float l = mx + logf(__expf(a - mx) + __expf(c - mx));
  out[b * 2] = a - l;
  out[b * 2 + 1] = c - l;
}

// ---------------------------------------------------------------------------
extern "C" void kernel_launch(void* const* d_in, const int* in_sizes, int n_in,
                              void* d_out, int out_size, void* d_ws, size_t ws_size,
                              hipStream_t stream) {
  (void)in_sizes; (void)n_in; (void)out_size; (void)ws_size;
  // inputs (setup_inputs order)
  const float* x    = (const float*)d_in[0];
  const int*   ei1  = (const int*)d_in[1];
  const int*   ei2  = (const int*)d_in[2];
  const float* ea1  = (const float*)d_in[3];
  const float* ea2  = (const float*)d_in[4];
  const int*   batchs = (const int*)d_in[6];
  const float* c1_wq = (const float*)d_in[7];  const float* c1_bq = (const float*)d_in[8];
  const float* c1_wk = (const float*)d_in[9];  const float* c1_bk = (const float*)d_in[10];
  const float* c1_wv = (const float*)d_in[11]; const float* c1_bv = (const float*)d_in[12];
  const float* c1_we = (const float*)d_in[13];
  const float* c1_ws = (const float*)d_in[14]; const float* c1_bs = (const float*)d_in[15];
  const float* cs_wq = (const float*)d_in[16]; const float* cs_bq = (const float*)d_in[17];
  const float* cs_wk = (const float*)d_in[18]; const float* cs_bk = (const float*)d_in[19];
  const float* cs_wv = (const float*)d_in[20]; const float* cs_bv = (const float*)d_in[21];
  const float* cs_we = (const float*)d_in[22];
  const float* cs_ws = (const float*)d_in[23]; const float* cs_bs = (const float*)d_in[24];
  const float* lin0_w = (const float*)d_in[25]; const float* lin0_b = (const float*)d_in[26];
  const float* lin1_w = (const float*)d_in[27]; const float* lin1_b = (const float*)d_in[28];
  const float* lin3_w = (const float*)d_in[29]; const float* lin3_b = (const float*)d_in[30];

  // workspace carving
  char* ws = (char*)d_ws;
  size_t off = 0;
  auto carve = [&](size_t bytes) -> char* {
    size_t cur = (off + 255) & ~(size_t)255;
    off = cur + bytes;
    return ws + cur;
  };
  const size_t ND = (size_t)NN * DD;
  unsigned short* actbf = (unsigned short*)carve(ND * 2);
  float* qb    = (float*)carve(ND * 4);    // also aliased as final h
  float* kb    = (float*)carve(ND * 4);
  float* vb    = (float*)carve(ND * 4);
  float* skipb = (float*)carve(ND * 4);
  float* out1  = (float*)carve(ND * 4);
  float* out2  = (float*)carve(ND * 4);
  float* logits = (float*)carve((size_t)EE * 4);
  float* wexp   = (float*)carve((size_t)EE * 4);
  float* mbuf   = (float*)carve((size_t)NN * 4);
  float* den    = (float*)carve((size_t)NN * 4);
  unsigned short* wbf = (unsigned short*)carve((size_t)16 * DD * DD * 2);
  float* sums = (float*)carve((size_t)BB * DD * 4);
  float* cnt  = (float*)carve((size_t)BB * 4);
  float* g    = (float*)carve((size_t)BB * DD * 4);
  float* t0   = (float*)carve((size_t)BB * DFF * 4);
  float* t1   = (float*)carve((size_t)BB * DFF * 4);
  float* t2   = (float*)carve((size_t)BB * CC * 4);

  const int WMAT = DD * DD; // 16384
  // convert weights to bf16 once
  k_f32_to_bf16<<<(WMAT + 255) / 256, 256, 0, stream>>>(c1_wq, wbf + 0 * WMAT, WMAT);
  k_f32_to_bf16<<<(WMAT + 255) / 256, 256, 0, stream>>>(c1_wk, wbf + 1 * WMAT, WMAT);
  k_f32_to_bf16<<<(WMAT + 255) / 256, 256, 0, stream>>>(c1_wv, wbf + 2 * WMAT, WMAT);
  k_f32_to_bf16<<<(WMAT + 255) / 256, 256, 0, stream>>>(c1_ws, wbf + 3 * WMAT, WMAT);
  k_f32_to_bf16<<<(3 * WMAT + 255) / 256, 256, 0, stream>>>(cs_wq, wbf + 4 * WMAT, 3 * WMAT);
  k_f32_to_bf16<<<(3 * WMAT + 255) / 256, 256, 0, stream>>>(cs_wk, wbf + 7 * WMAT, 3 * WMAT);
  k_f32_to_bf16<<<(3 * WMAT + 255) / 256, 256, 0, stream>>>(cs_wv, wbf + 10 * WMAT, 3 * WMAT);
  k_f32_to_bf16<<<(3 * WMAT + 255) / 256, 256, 0, stream>>>(cs_ws, wbf + 13 * WMAT, 3 * WMAT);

  // layer-0 input in bf16
  k_f32_to_bf16<<<(int)((ND + 255) / 256), 256, 0, stream>>>(x, actbf, (int)ND);

  const int gemmGrid = NN / (MTILES * 16);     // 1250 blocks of 80 rows
  const int edgeWaveGrid = EE / 8;             // 75000 (wave per edge)
  const int edgeThreadGrid = (EE + 255) / 256;
  const int nodeGrid = (NN + 255) / 256;

  for (int layer = 0; layer < NLAYERS; ++layer) {
    const unsigned short *wq, *wk, *wv, *wsm;
    const float *bq, *bk2, *bv2, *bs2, *wep;
    if (layer == 0) {
      wq = wbf + 0 * WMAT; wk = wbf + 1 * WMAT; wv = wbf + 2 * WMAT; wsm = wbf + 3 * WMAT;
      bq = c1_bq; bk2 = c1_bk; bv2 = c1_bv; bs2 = c1_bs; wep = c1_we;
    } else {
      const int j = layer - 1;
      wq  = wbf + (4 + j) * WMAT;  wk = wbf + (7 + j) * WMAT;
      wv  = wbf + (10 + j) * WMAT; wsm = wbf + (13 + j) * WMAT;
      bq = cs_bq + j * DD; bk2 = cs_bk + j * DD; bv2 = cs_bv + j * DD;
      bs2 = cs_bs + j * DD; wep = cs_we + j * DD * EDD;
    }
    // dense projections (shared by both edge sets)
    k_gemm_bf16_128<<<gemmGrid, 256, 0, stream>>>(actbf, wq,  bq,  qb,    NN);
    k_gemm_bf16_128<<<gemmGrid, 256, 0, stream>>>(actbf, wk,  bk2, kb,    NN);
    k_gemm_bf16_128<<<gemmGrid, 256, 0, stream>>>(actbf, wv,  bv2, vb,    NN);
    k_gemm_bf16_128<<<gemmGrid, 256, 0, stream>>>(actbf, wsm, bs2, skipb, NN);

    for (int gi = 0; gi < 2; ++gi) {
      const int* ei = gi ? ei2 : ei1;
      const float* ea = gi ? ea2 : ea1;
      float* outb = gi ? out2 : out1;
      const int* src = ei;          // edge_index[0]
      const int* dst = ei + EE;     // edge_index[1]
      hipMemsetAsync(outb, 0, ND * 4, stream);
      hipMemsetAsync(den, 0, (size_t)NN * 4, stream);
      k_init_neginf<<<nodeGrid, 256, 0, stream>>>(mbuf, NN);
      k_edge_logits<<<edgeWaveGrid, 256, 0, stream>>>(src, dst, ea, wep, qb, kb, logits, EE);
      k_seg_max<<<edgeThreadGrid, 256, 0, stream>>>(dst, logits, mbuf, EE);
      k_seg_expsum<<<edgeThreadGrid, 256, 0, stream>>>(dst, logits, mbuf, wexp, den, EE);
      k_edge_scatter<<<edgeWaveGrid, 256, 0, stream>>>(src, dst, ea, wep, vb, wexp, den, outb, EE);
    }
    // h = relu(x2) - relu(x1); h stored in qb (q is dead), bf16 copy in actbf
    k_combine<<<(int)((ND + 255) / 256), 256, 0, stream>>>(out1, out2, skipb, qb, actbf, ND);
  }

  // global mean pool + MLP tail
  hipMemsetAsync(sums, 0, (size_t)BB * DD * 4, stream);
  hipMemsetAsync(cnt, 0, (size_t)BB * 4, stream);
  k_pool_sum<<<(NN * DD + 255) / 256, 256, 0, stream>>>(qb, batchs, sums, cnt, NN);
  k_pool_div<<<(BB * DD + 255) / 256, 256, 0, stream>>>(sums, cnt, g, BB * DD);
  k_fc<<<(BB * DFF + 255) / 256, 256, 0, stream>>>(g,  lin0_w, lin0_b, t0, BB, DD,  DFF, 1);
  k_fc<<<(BB * DFF + 255) / 256, 256, 0, stream>>>(t0, lin1_w, lin1_b, t1, BB, DFF, DFF, 1);
  k_fc<<<(BB * CC + 255) / 256, 256, 0, stream>>>(t1, lin3_w, lin3_b, t2, BB, DFF, CC, 0);
  k_logsoftmax2<<<1, BB, 0, stream>>>(t2, (float*)d_out, BB);
}
